// TransformerBlock_63840393888272
// MI455X (gfx1250) — compile-verified, matrix operand order checked by reading, not run on hardware
//
#include <hip/hip_runtime.h>

// ---------------------------------------------------------------------------
// Types for CDNA5 WMMA (gfx1250, wave32)
// ---------------------------------------------------------------------------
typedef __bf16 bf16_t;
typedef __attribute__((ext_vector_type(16))) __bf16 v16bf;
typedef __attribute__((ext_vector_type(8)))  __bf16 v8bf;
typedef __attribute__((ext_vector_type(8)))  float  v8f;

#define BB 4
#define NN 4096
#define KNB 16
#define DP 128
#define DM 512

__device__ __forceinline__ bf16_t f2bf(float f) {
  unsigned u = __builtin_bit_cast(unsigned, f);
  u += 0x7FFFu + ((u >> 16) & 1u);              // round-to-nearest-even
  return __builtin_bit_cast(bf16_t, (unsigned short)(u >> 16));
}
__device__ __forceinline__ float bf2f(bf16_t h) {
  unsigned short s = __builtin_bit_cast(unsigned short, h);
  unsigned u = ((unsigned)s) << 16;
  return __builtin_bit_cast(float, u);
}
__device__ __forceinline__ v8f splat8(float x) {
  v8f v;
#pragma unroll
  for (int j = 0; j < 8; ++j) v[j] = x;
  return v;
}

// Async global->LDS copy, 16B per lane, tracked by ASYNCcnt (CDNA5).
__device__ __forceinline__ void async_gload_lds_b128(void* lds_ptr,
                                                     const void* gptr) {
  unsigned loff = (unsigned)(unsigned long long)lds_ptr;  // LDS byte offset
  asm volatile("global_load_async_to_lds_b128 %0, %1, off"
               :: "v"(loff), "v"(gptr)
               : "memory");
}
__device__ __forceinline__ void wait_async0() {
  asm volatile("s_wait_asynccnt 0x0" ::: "memory");
}

__device__ __forceinline__ v8f wmma_bf(v16bf a, v16bf b, v8f c) {
  return __builtin_amdgcn_wmma_f32_16x16x32_bf16(
      /*neg_a=*/false, a, /*neg_b=*/false, b,
      /*c_mod=*/(short)0, c, /*reuse_a=*/false, /*reuse_b=*/false);
}

// A-fragment for V_WMMA_F32_16X16X32_BF16 from a row-major [16 x stride] bf16
// matrix. ISA layout: lanes 0-15 -> M=lane, K = {0..7, 16..23};
//                     lanes 16-31 -> M=lane-16, K = {8..15, 24..31}.
__device__ __forceinline__ v16bf load_a_frag(const bf16_t* A, int stride,
                                             int kbase, int lane) {
  const int m    = lane & 15;
  const int kofs = (lane >= 16) ? 8 : 0;
  const bf16_t* p = A + (size_t)m * stride + kbase + kofs;
  v8bf lo = *(const v8bf*)(p);        // K = kofs + 0..7
  v8bf hi = *(const v8bf*)(p + 16);   // K = kofs + 16..23
  v16bf r;
#pragma unroll
  for (int j = 0; j < 8; ++j) { r[j] = lo[j]; r[j + 8] = hi[j]; }
  return r;
}

// 4-tile GEMM for one wave: one A-fragment load feeds 4 WMMAs (tiles
// nt0..nt0+3).  Wpacked holds per-(nt,kb) fragments of 32 lanes x 16 bf16 in
// exact B-fragment lane order (one contiguous 32B load per lane).
__device__ __forceinline__ void gemm4(const bf16_t* __restrict__ A, int Kdim,
                                      const bf16_t* __restrict__ Wp, int nt0,
                                      int lane, v8f& c0, v8f& c1, v8f& c2,
                                      v8f& c3) {
  const int kblocks = Kdim >> 5;
  const v16bf* bp = reinterpret_cast<const v16bf*>(Wp) + lane;
  const v16bf* s0 = bp + (size_t)((nt0 + 0) * kblocks) * 32;
  const v16bf* s1 = bp + (size_t)((nt0 + 1) * kblocks) * 32;
  const v16bf* s2 = bp + (size_t)((nt0 + 2) * kblocks) * 32;
  const v16bf* s3 = bp + (size_t)((nt0 + 3) * kblocks) * 32;
  for (int kb = 0; kb < kblocks; ++kb) {
    if (kb + 1 < kblocks) {
      __builtin_prefetch(s0 + (kb + 1) * 32, 0, 1);
      __builtin_prefetch(s1 + (kb + 1) * 32, 0, 1);
      __builtin_prefetch(s2 + (kb + 1) * 32, 0, 1);
      __builtin_prefetch(s3 + (kb + 1) * 32, 0, 1);
    }
    v16bf a = load_a_frag(A, Kdim, kb << 5, lane);
    c0 = wmma_bf(a, s0[kb * 32], c0);
    c1 = wmma_bf(a, s1[kb * 32], c1);
    c2 = wmma_bf(a, s2[kb * 32], c2);
    c3 = wmma_bf(a, s3[kb * 32], c3);
  }
}

// ---------------------------------------------------------------------------
// Kernel 1: KNN (top-16 smallest squared distance per query, includes self)
// ---------------------------------------------------------------------------
__global__ __launch_bounds__(256) void knn_kernel(const float* __restrict__ xyz,
                                                  int* __restrict__ knn) {
  __shared__ float sX[256], sY[256], sZ[256];
  const int t = threadIdx.x;
  const int q = blockIdx.x * 256 + t;      // global point id
  const int b = q >> 12;                   // /4096
  const float qx = xyz[(size_t)q * 3 + 0];
  const float qy = xyz[(size_t)q * 3 + 1];
  const float qz = xyz[(size_t)q * 3 + 2];

  float bd[KNB];
  int   bi[KNB];
#pragma unroll
  for (int s = 0; s < KNB; ++s) { bd[s] = 3.4e38f; bi[s] = 0; }

  for (int tile = 0; tile < NN / 256; ++tile) {
    const size_t p = (size_t)(b * NN + tile * 256 + t) * 3;
    sX[t] = xyz[p + 0]; sY[t] = xyz[p + 1]; sZ[t] = xyz[p + 2];
    __syncthreads();
    for (int j = 0; j < 256; ++j) {
      const float dx = qx - sX[j], dy = qy - sY[j], dz = qz - sZ[j];
      const float d = dx * dx + dy * dy + dz * dz;
      const int id = tile * 256 + j;
      if (d < bd[KNB - 1]) {
        // shift everything strictly greater than d one slot down
#pragma unroll
        for (int s = KNB - 1; s > 0; --s) {
          const bool mv = d < bd[s - 1];
          bd[s] = mv ? bd[s - 1] : bd[s];
          bi[s] = mv ? bi[s - 1] : bi[s];
        }
        // place d at the (unique) open slot, scanning high->low
#pragma unroll
        for (int s = KNB - 1; s >= 0; --s) {
          const bool pl = (d < bd[s]) && (s == 0 || d >= bd[s - 1]);
          if (pl) { bd[s] = d; bi[s] = id; }
        }
      }
    }
    __syncthreads();
  }
#pragma unroll
  for (int s = 0; s < KNB; ++s) knn[(size_t)q * KNB + s] = bi[s];
}

// ---------------------------------------------------------------------------
// Kernel 2: pack W[Kdim][Ndim] (f32 row-major) -> bf16 WMMA B-fragment order
// ---------------------------------------------------------------------------
__global__ __launch_bounds__(256) void pack_w(const float* __restrict__ W,
                                              bf16_t* __restrict__ out,
                                              int Kdim, int Ndim) {
  const int t = blockIdx.x * 256 + threadIdx.x;
  if (t >= Kdim * Ndim) return;
  const int frag = t >> 9;          // 512 bf16 per fragment
  const int r    = t & 511;
  const int lane = r >> 4;
  const int j    = r & 15;
  const int kblocks = Kdim >> 5;
  const int nt = frag / kblocks;
  const int kb = frag % kblocks;
  const int kofs = (lane >= 16) ? 8 : 0;
  const int kk = kb * 32 + kofs + (j < 8 ? j : j + 8);
  const int nn = nt * 16 + (lane & 15);
  out[t] = f2bf(W[(size_t)kk * Ndim + nn]);
}

// ---------------------------------------------------------------------------
// Kernel 3: x = features@fc1 + b; q = x@wq (f32); k = x@wk (bf16); v = x@wv
// One block = one 16-row M tile.  8 waves x 4 N-tiles = 512 columns.
// ---------------------------------------------------------------------------
__global__ __launch_bounds__(256) void gemm_qkv_kernel(
    const float* __restrict__ features, const float* __restrict__ fc1_b,
    const bf16_t* __restrict__ fc1p, const bf16_t* __restrict__ wqp,
    const bf16_t* __restrict__ wkp, const bf16_t* __restrict__ wvp,
    float* __restrict__ qbuf, bf16_t* __restrict__ kbuf,
    float* __restrict__ vbuf) {
  __shared__ __align__(32) bf16_t sF[16][DP];
  __shared__ __align__(32) bf16_t sX[16][DM];
  const int t = threadIdx.x;
  const int lane = t & 31, wave = t >> 5;
  const int r0 = blockIdx.x * 16;
  const int mb = (lane >= 16) ? 8 : 0;
  const int nt0 = wave * 4;
  const int c0 = (nt0 + 0) * 16 + (lane & 15);
  const int c1 = (nt0 + 1) * 16 + (lane & 15);
  const int c2 = (nt0 + 2) * 16 + (lane & 15);
  const int c3 = (nt0 + 3) * 16 + (lane & 15);

#pragma unroll
  for (int i = 0; i < 8; ++i) {          // 16*128 elems / 256 threads
    const int o = t + i * 256;
    const int m = o >> 7, c = o & 127;
    sF[m][c] = f2bf(features[(size_t)(r0 + m) * DP + c]);
  }
  __syncthreads();

  {
    v8f a0 = splat8(fc1_b[c0]), a1 = splat8(fc1_b[c1]);
    v8f a2 = splat8(fc1_b[c2]), a3 = splat8(fc1_b[c3]);
    gemm4(&sF[0][0], DP, fc1p, nt0, lane, a0, a1, a2, a3);
#pragma unroll
    for (int i = 0; i < 8; ++i) {
      sX[i + mb][c0] = f2bf(a0[i]);
      sX[i + mb][c1] = f2bf(a1[i]);
      sX[i + mb][c2] = f2bf(a2[i]);
      sX[i + mb][c3] = f2bf(a3[i]);
    }
  }
  __syncthreads();

  {
    v8f a0 = splat8(0.f), a1 = splat8(0.f), a2 = splat8(0.f), a3 = splat8(0.f);
    gemm4(&sX[0][0], DM, wqp, nt0, lane, a0, a1, a2, a3);
#pragma unroll
    for (int i = 0; i < 8; ++i) {
      const size_t row = (size_t)(r0 + i + mb);
      qbuf[row * DM + c0] = a0[i];
      qbuf[row * DM + c1] = a1[i];
      qbuf[row * DM + c2] = a2[i];
      qbuf[row * DM + c3] = a3[i];
    }
  }
  {
    v8f a0 = splat8(0.f), a1 = splat8(0.f), a2 = splat8(0.f), a3 = splat8(0.f);
    gemm4(&sX[0][0], DM, wkp, nt0, lane, a0, a1, a2, a3);
#pragma unroll
    for (int i = 0; i < 8; ++i) {
      const size_t row = (size_t)(r0 + i + mb);
      kbuf[row * DM + c0] = f2bf(a0[i]);
      kbuf[row * DM + c1] = f2bf(a1[i]);
      kbuf[row * DM + c2] = f2bf(a2[i]);
      kbuf[row * DM + c3] = f2bf(a3[i]);
    }
  }
  {
    v8f a0 = splat8(0.f), a1 = splat8(0.f), a2 = splat8(0.f), a3 = splat8(0.f);
    gemm4(&sX[0][0], DM, wvp, nt0, lane, a0, a1, a2, a3);
#pragma unroll
    for (int i = 0; i < 8; ++i) {
      const size_t row = (size_t)(r0 + i + mb);
      vbuf[row * DM + c0] = a0[i];
      vbuf[row * DM + c1] = a1[i];
      vbuf[row * DM + c2] = a2[i];
      vbuf[row * DM + c3] = a3[i];
    }
  }
}

// ---------------------------------------------------------------------------
// Kernel 4: fused per-query attention.  One block per (b, n).
// Gathered K rows are async-DMA'd into LDS (sB) while the pos-enc MLP+GEMM
// runs; ASYNCcnt is drained before sB is consumed.
// ---------------------------------------------------------------------------
__global__ __launch_bounds__(256) void attn_kernel(
    const float* __restrict__ xyz, const float* __restrict__ features,
    const float* __restrict__ d1_w, const float* __restrict__ d1_b,
    const bf16_t* __restrict__ d2p, const float* __restrict__ d2_b,
    const bf16_t* __restrict__ g1p, const float* __restrict__ g1_b,
    const bf16_t* __restrict__ g2p, const float* __restrict__ g2_b,
    const float* __restrict__ fc2_w, const float* __restrict__ fc2_b,
    const int* __restrict__ knn, const float* __restrict__ qbuf,
    const bf16_t* __restrict__ kbuf, const float* __restrict__ vbuf,
    float* __restrict__ out_res, float* __restrict__ out_attn) {
  __shared__ __align__(32) bf16_t sA[16][DM];   // pos-hidden, then hidden2
  __shared__ __align__(32) bf16_t sB[16][DM];   // k-stage, attn in, attn
  __shared__ __align__(32) bf16_t sPE[16][DM];  // pos_enc
  __shared__ float sRes[DM];
  __shared__ int   sIdx[16];
  __shared__ float sDx[16], sDy[16], sDz[16];

  const int bn = blockIdx.x;
  const int b  = bn >> 12;
  const int t  = threadIdx.x;
  const int lane = t & 31, wave = t >> 5;
  const int mb = (lane >= 16) ? 8 : 0;
  const int nt0 = wave * 4;
  const int c0 = (nt0 + 0) * 16 + (lane & 15);
  const int c1 = (nt0 + 1) * 16 + (lane & 15);
  const int c2 = (nt0 + 2) * 16 + (lane & 15);
  const int c3 = (nt0 + 3) * 16 + (lane & 15);

  if (t < 16) {
    const int id = knn[(size_t)bn * KNB + t];
    const size_t gp = (size_t)(b * NN + id) * 3;
    sIdx[t] = b * NN + id;
    sDx[t] = xyz[(size_t)bn * 3 + 0] - xyz[gp + 0];
    sDy[t] = xyz[(size_t)bn * 3 + 1] - xyz[gp + 1];
    sDz[t] = xyz[(size_t)bn * 3 + 2] - xyz[gp + 2];
  }
  __syncthreads();

  // async-stage the 16 gathered K rows (16 x 512 bf16 = 16KB) into sB;
  // overlaps with steps 1-2 below.
#pragma unroll
  for (int i = 0; i < 4; ++i) {
    const int chunk = t + i * 256;        // 1024 x 16B chunks
    const int m  = chunk >> 6;            // neighbor row
    const int ei = (chunk & 63) * 8;      // element offset (8 bf16 per 16B)
    async_gload_lds_b128(&sB[m][ei], kbuf + (size_t)sIdx[m] * DM + ei);
  }

  // step 1: pos hidden = relu(delta @ d1_w + d1_b)  (K=3, VALU)
  for (int i = 0; i < 32; ++i) {
    const int o = t + i * 256;
    const int m = o >> 9, c = o & 511;
    const float h = d1_b[c] + sDx[m] * d1_w[c] + sDy[m] * d1_w[DM + c] +
                    sDz[m] * d1_w[2 * DM + c];
    sA[m][c] = f2bf(fmaxf(h, 0.f));
  }
  __syncthreads();

  // step 2: pos_enc = sA @ d2 + d2_b  -> sPE (bf16)
  {
    v8f a0 = splat8(d2_b[c0]), a1 = splat8(d2_b[c1]);
    v8f a2 = splat8(d2_b[c2]), a3 = splat8(d2_b[c3]);
    gemm4(&sA[0][0], DM, d2p, nt0, lane, a0, a1, a2, a3);
#pragma unroll
    for (int i = 0; i < 8; ++i) {
      sPE[i + mb][c0] = f2bf(a0[i]);
      sPE[i + mb][c1] = f2bf(a1[i]);
      sPE[i + mb][c2] = f2bf(a2[i]);
      sPE[i + mb][c3] = f2bf(a3[i]);
    }
  }
  wait_async0();          // gathered K rows landed in sB
  __syncthreads();

  // step 3 (in place): sB = q - k_gathered + pos_enc  (bf16)
  for (int i = 0; i < 32; ++i) {
    const int o = t + i * 256;
    const int m = o >> 9, c = o & 511;
    const float a = qbuf[(size_t)bn * DM + c] - bf2f(sB[m][c]) + bf2f(sPE[m][c]);
    sB[m][c] = f2bf(a);
  }
  __syncthreads();

  // step 4: hidden2 = relu(sB @ g1 + g1_b) -> sA
  {
    v8f a0 = splat8(g1_b[c0]), a1 = splat8(g1_b[c1]);
    v8f a2 = splat8(g1_b[c2]), a3 = splat8(g1_b[c3]);
    gemm4(&sB[0][0], DM, g1p, nt0, lane, a0, a1, a2, a3);
#pragma unroll
    for (int i = 0; i < 8; ++i) {
      sA[i + mb][c0] = f2bf(fmaxf(a0[i], 0.f));
      sA[i + mb][c1] = f2bf(fmaxf(a1[i], 0.f));
      sA[i + mb][c2] = f2bf(fmaxf(a2[i], 0.f));
      sA[i + mb][c3] = f2bf(fmaxf(a3[i], 0.f));
    }
  }
  __syncthreads();

  // step 5: logits = (sA @ g2 + g2_b) / sqrt(DM) -> sB
  const float inv_sqrt_dm = 0.04419417382415922f;
  {
    v8f a0 = splat8(g2_b[c0]), a1 = splat8(g2_b[c1]);
    v8f a2 = splat8(g2_b[c2]), a3 = splat8(g2_b[c3]);
    gemm4(&sA[0][0], DM, g2p, nt0, lane, a0, a1, a2, a3);
#pragma unroll
    for (int i = 0; i < 8; ++i) {
      sB[i + mb][c0] = f2bf(a0[i] * inv_sqrt_dm);
      sB[i + mb][c1] = f2bf(a1[i] * inv_sqrt_dm);
      sB[i + mb][c2] = f2bf(a2[i] * inv_sqrt_dm);
      sB[i + mb][c3] = f2bf(a3[i] * inv_sqrt_dm);
    }
  }
  __syncthreads();

  // step 6: softmax over the 16 neighbors, per channel (2 channels/thread)
  for (int ci = 0; ci < 2; ++ci) {
    const int c = t * 2 + ci;
    float mx = -3.4e38f;
#pragma unroll
    for (int m = 0; m < 16; ++m) mx = fmaxf(mx, bf2f(sB[m][c]));
    float e[16];
    float s = 0.f;
#pragma unroll
    for (int m = 0; m < 16; ++m) { e[m] = __expf(bf2f(sB[m][c]) - mx); s += e[m]; }
    const float r = 1.f / s;
#pragma unroll
    for (int m = 0; m < 16; ++m) sB[m][c] = f2bf(e[m] * r);
  }
  __syncthreads();

  // step 7: attn output + res = sum_m attn * (v + pos_enc)
  for (int ci = 0; ci < 2; ++ci) {
    const int c = t * 2 + ci;
    float r = 0.f;
#pragma unroll
    for (int m = 0; m < 16; ++m) {
      const float a = bf2f(sB[m][c]);
      out_attn[((size_t)bn * KNB + m) * DM + c] = a;
      r += a * (vbuf[(size_t)sIdx[m] * DM + c] + bf2f(sPE[m][c]));
    }
    sRes[c] = r;
  }
  __syncthreads();

  // step 8: out = res @ fc2 + fc2_b + features   (512 -> 128)
  if (t < DP) {
    float o = fc2_b[t] + features[(size_t)bn * DP + t];
    for (int c = 0; c < DM; ++c) o += sRes[c] * fc2_w[(size_t)c * DP + t];
    out_res[(size_t)bn * DP + t] = o;
  }
}

// ---------------------------------------------------------------------------
// Launch
// ---------------------------------------------------------------------------
extern "C" void kernel_launch(void* const* d_in, const int* in_sizes, int n_in,
                              void* d_out, int out_size, void* d_ws,
                              size_t ws_size, hipStream_t stream) {
  (void)in_sizes; (void)n_in; (void)out_size; (void)ws_size;
  const float* xyz   = (const float*)d_in[0];
  const float* feat  = (const float*)d_in[1];
  const float* fc1_w = (const float*)d_in[2];
  const float* fc1_b = (const float*)d_in[3];
  const float* fc2_w = (const float*)d_in[4];
  const float* fc2_b = (const float*)d_in[5];
  const float* wq    = (const float*)d_in[6];
  const float* wk    = (const float*)d_in[7];
  const float* wv    = (const float*)d_in[8];
  const float* d1_w  = (const float*)d_in[9];
  const float* d1_b  = (const float*)d_in[10];
  const float* d2_w  = (const float*)d_in[11];
  const float* d2_b  = (const float*)d_in[12];
  const float* g1_w  = (const float*)d_in[13];
  const float* g1_b  = (const float*)d_in[14];
  const float* g2_w  = (const float*)d_in[15];
  const float* g2_b  = (const float*)d_in[16];

  const size_t ROWS = (size_t)BB * NN;   // 16384
  char* ws = (char*)d_ws;
  size_t off = 0;
  auto take = [&](size_t bytes) {
    char* p = ws + off;
    off = (off + bytes + 255) & ~(size_t)255;
    return p;
  };
  int*    knn  = (int*)   take(ROWS * KNB * 4);
  bf16_t* fc1p = (bf16_t*)take((size_t)DP * DM * 2);
  bf16_t* wqp  = (bf16_t*)take((size_t)DM * DM * 2);
  bf16_t* wkp  = (bf16_t*)take((size_t)DM * DM * 2);
  bf16_t* wvp  = (bf16_t*)take((size_t)DM * DM * 2);
  bf16_t* d2p  = (bf16_t*)take((size_t)DM * DM * 2);
  bf16_t* g1p  = (bf16_t*)take((size_t)DM * DM * 2);
  bf16_t* g2p  = (bf16_t*)take((size_t)DM * DM * 2);
  float*  qbuf = (float*) take(ROWS * DM * 4);
  bf16_t* kbuf = (bf16_t*)take(ROWS * DM * 2);
  float*  vbuf = (float*) take(ROWS * DM * 4);

  float* out_res  = (float*)d_out;
  float* out_attn = out_res + ROWS * DP;

  knn_kernel<<<(int)(ROWS / 256), 256, 0, stream>>>(xyz, knn);

  pack_w<<<(DP * DM + 255) / 256, 256, 0, stream>>>(fc1_w, fc1p, DP, DM);
  pack_w<<<(DM * DM + 255) / 256, 256, 0, stream>>>(wq, wqp, DM, DM);
  pack_w<<<(DM * DM + 255) / 256, 256, 0, stream>>>(wk, wkp, DM, DM);
  pack_w<<<(DM * DM + 255) / 256, 256, 0, stream>>>(wv, wvp, DM, DM);
  pack_w<<<(DM * DM + 255) / 256, 256, 0, stream>>>(d2_w, d2p, DM, DM);
  pack_w<<<(DM * DM + 255) / 256, 256, 0, stream>>>(g1_w, g1p, DM, DM);
  pack_w<<<(DM * DM + 255) / 256, 256, 0, stream>>>(g2_w, g2p, DM, DM);

  gemm_qkv_kernel<<<(int)(ROWS / 16), 256, 0, stream>>>(
      feat, fc1_b, fc1p, wqp, wkp, wvp, qbuf, kbuf, vbuf);

  attn_kernel<<<(int)ROWS, 256, 0, stream>>>(
      xyz, feat, d1_w, d1_b, d2p, d2_b, g1p, g1_b, g2p, g2_b, fc2_w, fc2_b,
      knn, qbuf, kbuf, vbuf, out_res, out_attn);
}
